// GATProcessor_28999619182945
// MI455X (gfx1250) — matrix operand order repaired and checked by reading, not compiled, not run on hardware
//
#include <hip/hip_runtime.h>

#define N_NODES 100000          /* divisible by 16: 6250 M-tiles */
#define N_EDGES 1600000
#define E_TOT   (N_EDGES + N_NODES)
#define HEADS   8
#define CH      32
#define HC      (HEADS * CH)    /* 256 */
#define MT_PER_WAVE 5           /* 6250 = 5 * 1250 -> exact blocking, no tail */

typedef __attribute__((ext_vector_type(16))) _Float16 v16h;
typedef __attribute__((ext_vector_type(8)))  _Float16 v8h;
typedef __attribute__((ext_vector_type(8)))  float    v8f;

// xl/xr are stored CHANNEL-MAJOR per node: element (node, c, h) at node*HC + c*HEADS + h.
// => a lane (= channel) reads all 8 head values as one contiguous 16B load.

__device__ __forceinline__ void edge_ends(const int* __restrict__ ei, int e, int& src, int& dst) {
    if (e < N_EDGES) { src = ei[e]; dst = ei[N_EDGES + e]; }
    else             { src = dst = e - N_EDGES; }        // appended self-loops
}

// float atomic max via signed/unsigned integer ordering trick (valid incl. -inf init)
__device__ __forceinline__ void atomicMaxF32(float* addr, float val) {
    if (val >= 0.0f) atomicMax((int*)addr, __float_as_int(val));
    else             atomicMin((unsigned int*)addr, __float_as_uint(val));
}

__global__ void fill_f32(float* __restrict__ p, float v, long n) {
    long i = (long)blockIdx.x * blockDim.x + threadIdx.x;
    if (i < n) p[i] = v;
}

// xl = x@Wl, xr = x@Wr.  K = CH = 32 -> one v_wmma_f32_16x16x32_f16 per 16x16 tile.
// One wave packs a 32x16 B tile once and reuses it across 5 M-tiles.  The M loop is
// deliberately NOT unrolled: one staged A tile at a time keeps VGPRs < spill point.
__global__ void gemm_dual_wmma(const float* __restrict__ x,
                               const float* __restrict__ Wl,
                               const float* __restrict__ Wr,
                               _Float16* __restrict__ xl,
                               _Float16* __restrict__ xr) {
    const int wave    = blockIdx.x * (blockDim.x >> 5) + (threadIdx.x >> 5);
    const int lane    = threadIdx.x & 31;
    const int mblocks = (N_NODES / 16) / MT_PER_WAVE;    // 1250
    if (wave >= mblocks * 32) return;
    const int  mblock = wave >> 5;
    const int  sub    = wave & 31;
    const int  which  = sub >> 4;          // 0 -> xl, 1 -> xr
    const int  ntile  = sub & 15;
    const bool hi     = (lane >= 16);
    const int  lrow   = lane & 15;

    // B: 32x16 f16 (lane n = col n; lanes 0-15 K=0..15, lanes 16-31 K=16..31)
    const float* W    = which ? Wr : Wl;
    const int    ncol = ntile * 16 + lrow;               // output column = h*CH + c
    const int    kb   = hi ? 16 : 0;
    v16h b;
#pragma unroll
    for (int i = 0; i < 16; ++i)
        b[i] = (_Float16)W[(long)(kb + i) * HC + ncol];

    _Float16* out = which ? xr : xl;
    // channel-major permutation of the output column
    const int ocol = (ncol & (CH - 1)) * HEADS + (ncol >> 5);   // c*8 + h

#pragma unroll 1
    for (int t = 0; t < MT_PER_WAVE; ++t) {
        const int mtile = mblock * MT_PER_WAVE + t;

        // A: 16x32 f16 (lanes 0-15 -> K{0..15}, lanes 16-31 -> K{16..31})
        // One contiguous 16-float run per lane -> 4x b128 loads.
        const int m = mtile * 16 + lrow;
        const float* xrow = x + (long)m * CH;
        const int k0 = hi ? 16 : 0;
        const float4 q0 = *(const float4*)(xrow + k0);
        const float4 q1 = *(const float4*)(xrow + k0 + 4);
        const float4 q2 = *(const float4*)(xrow + k0 + 8);
        const float4 q3 = *(const float4*)(xrow + k0 + 12);
        v16h a;
        a[0]  = (_Float16)q0.x; a[1]  = (_Float16)q0.y; a[2]  = (_Float16)q0.z; a[3]  = (_Float16)q0.w;
        a[4]  = (_Float16)q1.x; a[5]  = (_Float16)q1.y; a[6]  = (_Float16)q1.z; a[7]  = (_Float16)q1.w;
        a[8]  = (_Float16)q2.x; a[9]  = (_Float16)q2.y; a[10] = (_Float16)q2.z; a[11] = (_Float16)q2.w;
        a[12] = (_Float16)q3.x; a[13] = (_Float16)q3.y; a[14] = (_Float16)q3.z; a[15] = (_Float16)q3.w;

        v8f c = {};
        c = __builtin_amdgcn_wmma_f32_16x16x32_f16(false, a, false, b, (short)0, c, false, false);

        // C/D: VGPR r -> row r (lanes 0-15) or row r+8 (lanes 16-31)
        _Float16* orow = out + (long)(mtile * 16 + (hi ? 8 : 0)) * HC + ocol;
#pragma unroll
        for (int r = 0; r < 8; ++r)
            orow[(long)r * HC] = (_Float16)c[r];
    }
}

// One wave per edge, lane = channel; one b128 per node row (channel-major f16).
// Per head: leaky(xl[src]+xr[dst])*att, xor-tree reduce across channels.
__global__ void edge_logits_kernel(const _Float16* __restrict__ xl,
                                   const _Float16* __restrict__ xr,
                                   const int* __restrict__ ei,
                                   const float* __restrict__ att,
                                   float* __restrict__ pbuf,
                                   float* __restrict__ mmax) {
    const int e    = blockIdx.x * (blockDim.x >> 5) + (threadIdx.x >> 5);
    const int lane = threadIdx.x & 31;
    if (e >= E_TOT) return;
    int src, dst; edge_ends(ei, e, src, dst);
    const v8h hl = *(const v8h*)(xl + (long)src * HC + lane * HEADS);
    const v8h hr = *(const v8h*)(xr + (long)dst * HC + lane * HEADS);
#pragma unroll
    for (int h = 0; h < HEADS; ++h) {
        float v = (float)hl[h] + (float)hr[h];
        v = (v > 0.0f) ? v : 0.2f * v;             // GATv2 leaky slope 0.2
        v *= att[h * CH + lane];
#pragma unroll
        for (int off = 16; off >= 1; off >>= 1)
            v += __shfl_xor(v, off, 32);
        if (lane == 0) {
            pbuf[(long)e * HEADS + h] = v;
            atomicMaxF32(&mmax[dst * HEADS + h], v);
        }
    }
}

__global__ void softmax_num_kernel(const int* __restrict__ ei,
                                   float* __restrict__ pbuf,
                                   const float* __restrict__ mmax,
                                   float* __restrict__ denom) {
    long idx = (long)blockIdx.x * blockDim.x + threadIdx.x;
    if (idx >= (long)E_TOT * HEADS) return;
    int e = (int)(idx >> 3), h = (int)(idx & 7);
    int src, dst; edge_ends(ei, e, src, dst);
    float p = __expf(pbuf[idx] - mmax[dst * HEADS + h]);
    pbuf[idx] = p;
    atomicAdd(&denom[dst * HEADS + h], p);
}

// One wave per edge: lanes 0-7 compute alpha_h, broadcast via shfl; one b128 per row.
__global__ void aggregate_kernel(const _Float16* __restrict__ xl,
                                 const int* __restrict__ ei,
                                 const float* __restrict__ pbuf,
                                 const float* __restrict__ denom,
                                 float* __restrict__ agg) {
    const int e    = blockIdx.x * (blockDim.x >> 5) + (threadIdx.x >> 5);
    const int lane = threadIdx.x & 31;
    if (e >= E_TOT) return;
    int src, dst; edge_ends(ei, e, src, dst);
    float aval = 0.0f;
    if (lane < HEADS)
        aval = pbuf[(long)e * HEADS + lane] / denom[dst * HEADS + lane];
    const v8h hl = *(const v8h*)(xl + (long)src * HC + lane * HEADS);
    float acc = 0.0f;
#pragma unroll
    for (int h = 0; h < HEADS; ++h)
        acc += __shfl(aval, h, 32) * (float)hl[h];
    atomicAdd(&agg[dst * CH + lane], acc * (1.0f / HEADS));
}

__global__ void post_residual_kernel(const float* __restrict__ agg,
                                     const float* __restrict__ bias,
                                     const float* __restrict__ x,
                                     float* __restrict__ y, int act) {
    long idx = (long)blockIdx.x * blockDim.x + threadIdx.x;
    if (idx >= (long)N_NODES * CH) return;
    int c = (int)(idx & (CH - 1));
    float v = agg[idx] + bias[c];
    if (act) v = (v > 0.0f) ? v : 0.01f * v;       // outer leaky slope 0.01
    y[idx] = v + x[idx];
}

__global__ void bn_reduce_kernel(const float* __restrict__ y, float* __restrict__ sums) {
    __shared__ float ssum[CH], ssq[CH];
    if (threadIdx.x < CH) { ssum[threadIdx.x] = 0.0f; ssq[threadIdx.x] = 0.0f; }
    __syncthreads();
    const long total  = (long)N_NODES * CH;
    long idx = (long)blockIdx.x * blockDim.x + threadIdx.x;
    const long stride = (long)gridDim.x * blockDim.x;   // multiple of 32 -> column fixed
    float s = 0.0f, q = 0.0f;
    for (; idx < total; idx += stride) { float v = y[idx]; s += v; q += v * v; }
    const int c = threadIdx.x & (CH - 1);
    atomicAdd(&ssum[c], s);
    atomicAdd(&ssq[c], q);
    __syncthreads();
    if (threadIdx.x < CH) {
        atomicAdd(&sums[threadIdx.x],      ssum[threadIdx.x]);
        atomicAdd(&sums[CH + threadIdx.x], ssq[threadIdx.x]);
    }
}

__global__ void bn_normalize_kernel(const float* __restrict__ y,
                                    const float* __restrict__ sums,
                                    const float* __restrict__ gamma,
                                    const float* __restrict__ beta,
                                    float* __restrict__ xout) {
    long idx = (long)blockIdx.x * blockDim.x + threadIdx.x;
    if (idx >= (long)N_NODES * CH) return;
    int c = (int)(idx & (CH - 1));
    const float invn = 1.0f / (float)N_NODES;
    float mu  = sums[c] * invn;
    float var = sums[CH + c] * invn - mu * mu;
    float inv = rsqrtf(var + 1e-5f);
    xout[idx] = (y[idx] - mu) * inv * gamma[c] + beta[c];
}

extern "C" void kernel_launch(void* const* d_in, const int* in_sizes, int n_in,
                              void* d_out, int out_size, void* d_ws, size_t ws_size,
                              hipStream_t stream) {
    (void)in_sizes; (void)n_in; (void)out_size; (void)ws_size;
    const float* x0 = (const float*)d_in[0];
    const int*   ei = (const int*)d_in[1];       // [2, N_EDGES] int32
    // d_in[2] = edge_attr (unused by reference math)

    // Workspace layout (f16 for the L2-resident gather arrays, f32 elsewhere)
    _Float16* xl = (_Float16*)d_ws;                          // N*HC f16 = 51.2 MB
    _Float16* xr = xl + (long)N_NODES * HC;                  // 51.2 MB
    float* fbase = (float*)(xr + (long)N_NODES * HC);
    long o = 0;
    float* pbuf  = fbase + o; o += (long)E_TOT * HEADS;      // 54.4 MB
    float* mmax  = fbase + o; o += (long)N_NODES * HEADS;
    float* denom = fbase + o; o += (long)N_NODES * HEADS;
    float* agg   = fbase + o; o += (long)N_NODES * CH;
    float* ybuf  = fbase + o; o += (long)N_NODES * CH;
    float* xbuf  = fbase + o; o += (long)N_NODES * CH;
    float* sums  = fbase + o;                                // 64 floats

    const long nh = (long)N_NODES * HEADS;
    const long nc = (long)N_NODES * CH;
    const long eh = (long)E_TOT * HEADS;
    const int  mblocks = (N_NODES / 16) / MT_PER_WAVE;       // 1250, exact
    const int  gwaves  = mblocks * 32;                       // 40000 waves

    const float* xin = x0;
    for (int L = 0; L < 3; ++L) {
        const float* Wl    = (const float*)d_in[3 + 6 * L + 0];
        const float* Wr    = (const float*)d_in[3 + 6 * L + 1];
        const float* att   = (const float*)d_in[3 + 6 * L + 2];
        const float* bias  = (const float*)d_in[3 + 6 * L + 3];
        const float* gamma = (const float*)d_in[3 + 6 * L + 4];
        const float* beta  = (const float*)d_in[3 + 6 * L + 5];
        float* xout = (L == 2) ? (float*)d_out : xbuf;

        fill_f32<<<(int)((nh + 255) / 256), 256, 0, stream>>>(mmax, -3.0e38f, nh);
        fill_f32<<<(int)((nh + 255) / 256), 256, 0, stream>>>(denom, 0.0f, nh);
        fill_f32<<<(int)((nc + 255) / 256), 256, 0, stream>>>(agg, 0.0f, nc);
        fill_f32<<<1, 64, 0, stream>>>(sums, 0.0f, 64);

        gemm_dual_wmma<<<gwaves / 8, 256, 0, stream>>>(xin, Wl, Wr, xl, xr);
        edge_logits_kernel<<<(E_TOT + 7) / 8, 256, 0, stream>>>(xl, xr, ei, att, pbuf, mmax);
        softmax_num_kernel<<<(int)((eh + 255) / 256), 256, 0, stream>>>(ei, pbuf, mmax, denom);
        aggregate_kernel<<<(E_TOT + 7) / 8, 256, 0, stream>>>(xl, ei, pbuf, denom, agg);
        post_residual_kernel<<<(int)((nc + 255) / 256), 256, 0, stream>>>(agg, bias, xin, ybuf, (L < 2) ? 1 : 0);
        bn_reduce_kernel<<<1024, 256, 0, stream>>>(ybuf, sums);
        bn_normalize_kernel<<<(int)((nc + 255) / 256), 256, 0, stream>>>(ybuf, sums, gamma, beta, xout);

        xin = xout;
    }
}
